// MultiTaskGAT_1958505087787
// MI455X (gfx1250) — matrix operand order repaired and checked by reading, not compiled, not run on hardware
//
#include <hip/hip_runtime.h>
#include <hip/hip_bf16.h>
#include <math.h>

// ---------------------------------------------------------------------------
// MultiTaskGAT on MI455X (gfx1250): bf16 WMMA GEMMs + atomic segment softmax.
//
//   N_NODES=40000, N_EDGES=400000, N_GRAPHS=128, IN_F=768, HID=128, HEADS=4
//
// GEMM strategy: A (node features) streams from HBM as f32 with on-the-fly
// v_cvt_pk_bf16_f32; B (weights, L2-resident, reused 2500x) is pre-packed
// once into WMMA B-fragment order so each fragment is ONE contiguous 32-byte
// per-lane load. Each wave holds 4 f32 accumulators (16x64 tile) so one A
// fragment feeds 4 v_wmma_f32_16x16x32_bf16 ops per K-step.
// ---------------------------------------------------------------------------

#define N_NODES   40000
#define N_EDGES   400000
#define N_GRAPHS  128
#define IN_F      768
#define HID       128
#define HEADS     4
#define NEG_SLOPE 0.2f

typedef __attribute__((ext_vector_type(16))) __bf16 v16bf;
typedef __attribute__((ext_vector_type(8)))  float  v8f;

#define WMMA_BF16(a, b, c) \
    __builtin_amdgcn_wmma_f32_16x16x32_bf16(false, (a), false, (b), (short)0, (c), false, false)

// ---------------------------------------------------------------------------
// Pack B[K,N] (f32 row-major) into WMMA B-fragment order as bf16.
// Fragment layout (this code's contiguous k order, same as the A loads; wave32, B 32x16):
//   lanes 0-15 : col N = lane,    K = kt*32 + 0..15
//   lanes 16-31: col N = lane-16, K = kt*32 + 16..31
// Output: frag (kt, nt) at offset ((kt*(N/16)+nt)*32 + lane)*16 bf16 elems,
// i.e. each lane's 16 values are contiguous (32 bytes).
// ---------------------------------------------------------------------------
__global__ void pack_b_frag(const float* __restrict__ B, __bf16* __restrict__ out,
                            int K, int N)
{
    int t = blockIdx.x * blockDim.x + threadIdx.x;
    int total = (K / 32) * (N / 16) * 32;
    if (t >= total) return;
    int lane = t & 31;
    int tile = t >> 5;               // kt*(N/16) + nt
    int NT = N >> 4;
    int nt = tile % NT;
    int kt = tile / NT;
    int half = lane >> 4, r = lane & 15;
    const float* src = B + (size_t)(kt * 32 + half * 16) * N + nt * 16 + r;
    __bf16* dst = out + (size_t)t * 16;
#pragma unroll
    for (int j = 0; j < 16; ++j) dst[j] = (__bf16)src[(size_t)j * N];
}

// ---------------------------------------------------------------------------
// C[M,N] = A[M,K] (f32 row-major) @ Bp (pre-packed bf16 fragments)
// Block = 128 threads = 4 wave32s. gridDim.x = M/16.
// Wave w owns the 16x64 tile at cols [ (blockIdx.y*4 + w)*64, +64 ).
// Optional epilogue: +bias[col], ReLU.
// ---------------------------------------------------------------------------
__global__ void wmma_gemm_bpk(const float* __restrict__ A,
                              const __bf16* __restrict__ Bp,
                              float* __restrict__ C,
                              int M, int K, int N,
                              const float* __restrict__ bias, int do_relu)
{
    const int wave = threadIdx.x >> 5;
    const int lane = threadIdx.x & 31;
    const int half = lane >> 4;      // 0: lanes 0-15, 1: lanes 16-31
    const int r    = lane & 15;
    const int row0 = blockIdx.x * 16;
    const int col64 = (blockIdx.y * 4 + wave) * 64;
    if (col64 >= N) return;          // wave-uniform: EXEC stays all-ones
    const int NT  = N >> 4;          // number of 16-col tiles in B
    const int nt0 = col64 >> 4;      // first of this wave's 4 col tiles

    const float* __restrict__ arow = A + (size_t)(row0 + r) * K + half * 16;
    const __bf16* __restrict__ bbase =
        Bp + ((size_t)nt0 * 32 + lane) * 16;   // += kt*NT*32*16 per K-step

    v8f acc0 = {}, acc1 = {}, acc2 = {}, acc3 = {};
    const size_t bstride = (size_t)NT * 32 * 16;   // bf16 elems per K-step

    for (int k0 = 0; k0 < K; k0 += 32) {
        // ---- A fragment: 16 contiguous-ish f32 -> bf16 (2x b128 pairs) ----
        const float* ap = arow + k0;
        v16bf a;
#pragma unroll
        for (int j = 0; j < 8; ++j) {
            a[j]     = (__bf16)ap[j];
            a[8 + j] = (__bf16)ap[8 + j];
        }
        // ---- B fragments: 32B contiguous per lane, 4 col tiles ----
        const __bf16* bp = bbase + (size_t)(k0 >> 5) * bstride;
        v16bf b0 = *(const v16bf*)(bp);
        v16bf b1 = *(const v16bf*)(bp + 1 * 32 * 16);
        v16bf b2 = *(const v16bf*)(bp + 2 * 32 * 16);
        v16bf b3 = *(const v16bf*)(bp + 3 * 32 * 16);
        if (k0 + 32 < K) {
            __builtin_prefetch(ap + 32, 0, 0);            // next A K-tile
            __builtin_prefetch(bp + bstride, 0, 0);       // next B fragments
        }
        acc0 = WMMA_BF16(a, b0, acc0);
        acc1 = WMMA_BF16(a, b1, acc1);
        acc2 = WMMA_BF16(a, b2, acc2);
        acc3 = WMMA_BF16(a, b3, acc3);
    }

    // ---- epilogue / store: C tile layout: VGPR v -> M = v + 8*half ----
    const int mbase = row0 + half * 8;
#define STORE_TILE(ACC, S)                                                    \
    {                                                                         \
        const int col = col64 + (S) * 16 + r;                                 \
        const float bv = bias ? bias[col] : 0.0f;                             \
        _Pragma("unroll")                                                     \
        for (int v = 0; v < 8; ++v) {                                         \
            float val = (ACC)[v] + bv;                                        \
            if (do_relu) val = val > 0.0f ? val : 0.0f;                       \
            C[(size_t)(mbase + v) * N + col] = val;                           \
        }                                                                     \
    }
    STORE_TILE(acc0, 0)
    STORE_TILE(acc1, 1)
    STORE_TILE(acc2, 2)
    STORE_TILE(acc3, 3)
#undef STORE_TILE
}

// ---------------------------------------------------------------------------
// Small utility kernels
// ---------------------------------------------------------------------------
__global__ void fill_f32(float* __restrict__ p, float v, long long n)
{
    long long t = (long long)blockIdx.x * blockDim.x + threadIdx.x;
    if (t < n) p[t] = v;
}

__global__ void add_bias_inplace(float* __restrict__ o,
                                 const float* __restrict__ bias,
                                 long long total, int C)
{
    long long t = (long long)blockIdx.x * blockDim.x + threadIdx.x;
    if (t < total) o[t] += bias[(int)(t % C)];
}

__device__ __forceinline__ void atomicMaxF(float* addr, float val)
{
    // monotonic int encoding: works for mixed-sign floats with -inf init
    if (val >= 0.0f) {
        atomicMax((int*)addr, __float_as_int(val));
    } else {
        atomicMin((unsigned int*)addr, (unsigned int)__float_as_int(val));
    }
}

// ---------------------------------------------------------------------------
// Attention scores: el[n,h] = dot(feat[n,h,:], al[h,:]); er likewise.
// ---------------------------------------------------------------------------
__global__ void attn_scores(const float* __restrict__ feat,
                            const float* __restrict__ al,
                            const float* __restrict__ ar,
                            float* __restrict__ el, float* __restrict__ er,
                            int H)
{
    int t = blockIdx.x * blockDim.x + threadIdx.x;
    if (t >= N_NODES * H) return;
    int h = t % H, n = t / H;
    const float* f = feat + ((size_t)n * H + h) * HID;
    const float* a = al + (size_t)h * HID;
    const float* b = ar + (size_t)h * HID;
    float sl = 0.0f, sr = 0.0f;
#pragma unroll 4
    for (int i = 0; i < HID; i += 4) {
        float4 fv = *(const float4*)(f + i);
        float4 av = *(const float4*)(a + i);
        float4 bv = *(const float4*)(b + i);
        sl += fv.x * av.x + fv.y * av.y + fv.z * av.z + fv.w * av.w;
        sr += fv.x * bv.x + fv.y * bv.y + fv.z * bv.z + fv.w * bv.w;
    }
    el[t] = sl;
    er[t] = sr;
}

// pass A: per edge-head logit + segment max into emax[dst,h]
__global__ void edge_logits(const float* __restrict__ el,
                            const float* __restrict__ er,
                            const int* __restrict__ src,
                            const int* __restrict__ dst,
                            float* __restrict__ elog,
                            float* __restrict__ emax, int H)
{
    int t = blockIdx.x * blockDim.x + threadIdx.x;
    if (t >= N_EDGES * H) return;
    int h = t % H, e = t / H;
    float v = el[src[e] * H + h] + er[dst[e] * H + h];
    v = v > 0.0f ? v : NEG_SLOPE * v;
    elog[t] = v;
    atomicMaxF(&emax[dst[e] * H + h], v);
}

// pass B: ee = exp(e - emax[dst]); denom[dst,h] += ee
__global__ void edge_exp(const float* __restrict__ emax,
                         const int* __restrict__ dst,
                         float* __restrict__ elog,
                         float* __restrict__ denom, int H)
{
    int t = blockIdx.x * blockDim.x + threadIdx.x;
    if (t >= N_EDGES * H) return;
    int h = t % H, e = t / H;
    float ex = __expf(elog[t] - emax[dst[e] * H + h]);
    elog[t] = ex;
    atomicAdd(&denom[dst[e] * H + h], ex);
}

// pass C: out[dst,h,:] += feat[src,h,:] * (ee/denom[dst,h])
// one thread per (edge, head, 4-float feature group): E*H*32 threads
__global__ void edge_scatter(const float* __restrict__ feat,
                             const float* __restrict__ elog,
                             const float* __restrict__ denom,
                             const int* __restrict__ src,
                             const int* __restrict__ dst,
                             float* __restrict__ out, int H)
{
    long long t = (long long)blockIdx.x * blockDim.x + threadIdx.x;
    long long total = (long long)N_EDGES * H * (HID / 4);
    if (t >= total) return;
    int fg = (int)(t & 31);
    int eh = (int)(t >> 5);
    int h = eh % H, e = eh / H;
    int s = src[e], d = dst[e];
    float alpha = elog[eh] / denom[d * H + h];
    float4 fv = *(const float4*)(feat + ((size_t)s * H + h) * HID + fg * 4);
    float* o = out + ((size_t)d * H + h) * HID + fg * 4;
    atomicAdd(o + 0, fv.x * alpha);
    atomicAdd(o + 1, fv.y * alpha);
    atomicAdd(o + 2, fv.z * alpha);
    atomicAdd(o + 3, fv.w * alpha);
}

// ---------------------------------------------------------------------------
// Graph pooling: hg_sum[g,:] += h[n,:]; cnt[g] += 1
// ---------------------------------------------------------------------------
__global__ void graph_accum(const float* __restrict__ h,
                            const int* __restrict__ gid,
                            float* __restrict__ hg, float* __restrict__ cnt)
{
    int t = blockIdx.x * blockDim.x + threadIdx.x;
    if (t >= N_NODES * (HID / 4)) return;
    int fg = t & 31;
    int n = t >> 5;
    int g = gid[n];
    float4 v = *(const float4*)(h + (size_t)n * HID + fg * 4);
    float* o = hg + (size_t)g * HID + fg * 4;
    atomicAdd(o + 0, v.x);
    atomicAdd(o + 1, v.y);
    atomicAdd(o + 2, v.z);
    atomicAdd(o + 3, v.w);
    if (fg == 0) atomicAdd(&cnt[g], 1.0f);
}

__global__ void graph_div(float* __restrict__ hg, const float* __restrict__ cnt)
{
    int t = blockIdx.x * blockDim.x + threadIdx.x;
    if (t >= N_GRAPHS * HID) return;
    hg[t] /= fmaxf(cnt[t / HID], 1.0f);
}

// ---------------------------------------------------------------------------
// Output heads
// ---------------------------------------------------------------------------
__global__ void node_head(const float* __restrict__ tbuf,
                          const float* __restrict__ nW2,
                          const float* __restrict__ nb2,
                          float* __restrict__ out)
{
    int n = blockIdx.x * blockDim.x + threadIdx.x;
    if (n >= N_NODES) return;
    const float* h = tbuf + (size_t)n * HID;
    float s0 = nb2[0], s1 = nb2[1];
#pragma unroll 4
    for (int i = 0; i < HID; ++i) {
        float v = h[i];
        s0 += v * nW2[i * 2 + 0];
        s1 += v * nW2[i * 2 + 1];
    }
    out[n * 2 + 0] = s0;
    out[n * 2 + 1] = s1;
}

__global__ void graph_mlp(const float* __restrict__ hg,
                          const float* __restrict__ gW1,
                          const float* __restrict__ gb1,
                          const float* __restrict__ gW2,
                          const float* __restrict__ gb2,
                          float* __restrict__ out)
{
    int g = blockIdx.x;
    int f = threadIdx.x;  // 128 threads
    __shared__ float hid[HID];
    const float* h = hg + (size_t)g * HID;
    float s = gb1[f];
    for (int i = 0; i < HID; ++i) s += h[i] * gW1[i * HID + f];
    hid[f] = s > 0.0f ? s : 0.0f;
    __syncthreads();
    if (f < 2) {
        float s2 = gb2[f];
        for (int i = 0; i < HID; ++i) s2 += hid[i] * gW2[i * 2 + f];
        out[g * 2 + f] = s2;
    }
}

// ---------------------------------------------------------------------------
// Host-side orchestration
// ---------------------------------------------------------------------------
static inline int blocks_for(long long n, int bs) { return (int)((n + bs - 1) / bs); }

extern "C" void kernel_launch(void* const* d_in, const int* in_sizes, int n_in,
                              void* d_out, int out_size, void* d_ws, size_t ws_size,
                              hipStream_t stream)
{
    const float* x    = (const float*)d_in[0];
    const int*   src  = (const int*)d_in[1];
    const int*   dst  = (const int*)d_in[2];
    const int*   gid  = (const int*)d_in[3];
    const float* W1   = (const float*)d_in[4];
    const float* al1  = (const float*)d_in[5];
    const float* ar1  = (const float*)d_in[6];
    const float* b1   = (const float*)d_in[7];
    const float* W2   = (const float*)d_in[8];
    const float* al2  = (const float*)d_in[9];
    const float* ar2  = (const float*)d_in[10];
    const float* b2   = (const float*)d_in[11];
    const float* nW1  = (const float*)d_in[12];
    const float* nb1  = (const float*)d_in[13];
    const float* nW2  = (const float*)d_in[14];
    const float* nb2  = (const float*)d_in[15];
    const float* gW1  = (const float*)d_in[16];
    const float* gb1  = (const float*)d_in[17];
    const float* gW2  = (const float*)d_in[18];
    const float* gb2  = (const float*)d_in[19];
    float* out = (float*)d_out;

    // ---- workspace layout (floats; all region offsets 32B-aligned) ----
    float* ws = (float*)d_ws;
    const size_t SZ_F1 = (size_t)N_NODES * HEADS * HID;  // 20.48M
    float* F1   = ws;                 // feat1 [N,4,128]; later feat2 + MLP hidden
    float* O1   = F1 + SZ_F1;         // layer1 aggregated out [N,512]; later h2
    float* EL   = O1 + SZ_F1;         // [N, H]
    float* ER   = EL + (size_t)N_NODES * HEADS;
    float* EMAX = ER + (size_t)N_NODES * HEADS;
    float* EDEN = EMAX + (size_t)N_NODES * HEADS;
    float* ELOG = EDEN + (size_t)N_NODES * HEADS;          // [E, H]
    float* HG   = ELOG + (size_t)N_EDGES * HEADS;          // [G, 128]
    float* CNT  = HG + (size_t)N_GRAPHS * HID;             // [G]
    __bf16* PB1 = (__bf16*)(CNT + N_GRAPHS);               // packed W1 (768*512)
    __bf16* PB2 = PB1 + (size_t)IN_F * HEADS * HID;        // packed W2 (512*128)
    __bf16* PB3 = PB2 + (size_t)HEADS * HID * HID;         // packed nW1 (128*128)
    // aliases for later phases
    float* F2 = F1;                          // feat2 [N,128]
    float* T  = F1 + (size_t)N_NODES * HID;  // node-MLP hidden [N,128]
    float* O2 = O1;                          // h2 [N,128]

    const int BS = 256;

    // ---- pack all weight matrices into WMMA B-fragment order (bf16) ----
    {
        int t1 = (IN_F / 32) * ((HEADS * HID) / 16) * 32;
        pack_b_frag<<<blocks_for(t1, BS), BS, 0, stream>>>(W1, PB1, IN_F, HEADS * HID);
        int t2 = ((HEADS * HID) / 32) * (HID / 16) * 32;
        pack_b_frag<<<blocks_for(t2, BS), BS, 0, stream>>>(W2, PB2, HEADS * HID, HID);
        int t3 = (HID / 32) * (HID / 16) * 32;
        pack_b_frag<<<blocks_for(t3, BS), BS, 0, stream>>>(nW1, PB3, HID, HID);
    }

    // =========== Layer 1 ===========
    fill_f32<<<blocks_for((long long)SZ_F1, BS), BS, 0, stream>>>(O1, 0.0f, (long long)SZ_F1);
    fill_f32<<<blocks_for(N_NODES * HEADS, BS), BS, 0, stream>>>(EMAX, -INFINITY, N_NODES * HEADS);
    fill_f32<<<blocks_for(N_NODES * HEADS, BS), BS, 0, stream>>>(EDEN, 0.0f, N_NODES * HEADS);

    // feat1 = x @ W1   (40000x768 @ 768x512): grid.y covers 8 col64-tiles / 4 waves
    {
        dim3 grid(N_NODES / 16, (HEADS * HID) / 256);
        wmma_gemm_bpk<<<grid, 128, 0, stream>>>(x, PB1, F1, N_NODES, IN_F, HEADS * HID, nullptr, 0);
    }
    attn_scores<<<blocks_for(N_NODES * HEADS, BS), BS, 0, stream>>>(F1, al1, ar1, EL, ER, HEADS);
    edge_logits<<<blocks_for((long long)N_EDGES * HEADS, BS), BS, 0, stream>>>(EL, ER, src, dst, ELOG, EMAX, HEADS);
    edge_exp<<<blocks_for((long long)N_EDGES * HEADS, BS), BS, 0, stream>>>(EMAX, dst, ELOG, EDEN, HEADS);
    edge_scatter<<<blocks_for((long long)N_EDGES * HEADS * (HID / 4), BS), BS, 0, stream>>>(
        F1, ELOG, EDEN, src, dst, O1, HEADS);
    add_bias_inplace<<<blocks_for((long long)SZ_F1, BS), BS, 0, stream>>>(O1, b1, (long long)SZ_F1, HEADS * HID);

    // =========== Layer 2 ===========
    // feat2 = h1 @ W2   (40000x512 @ 512x128): 2 of 4 waves active per block
    {
        dim3 grid(N_NODES / 16, 1);
        wmma_gemm_bpk<<<grid, 128, 0, stream>>>(O1, PB2, F2, N_NODES, HEADS * HID, HID, nullptr, 0);
    }
    fill_f32<<<blocks_for((long long)N_NODES * HID, BS), BS, 0, stream>>>(O2, 0.0f, (long long)N_NODES * HID);
    fill_f32<<<blocks_for(N_NODES, BS), BS, 0, stream>>>(EMAX, -INFINITY, N_NODES);
    fill_f32<<<blocks_for(N_NODES, BS), BS, 0, stream>>>(EDEN, 0.0f, N_NODES);

    attn_scores<<<blocks_for(N_NODES, BS), BS, 0, stream>>>(F2, al2, ar2, EL, ER, 1);
    edge_logits<<<blocks_for((long long)N_EDGES, BS), BS, 0, stream>>>(EL, ER, src, dst, ELOG, EMAX, 1);
    edge_exp<<<blocks_for((long long)N_EDGES, BS), BS, 0, stream>>>(EMAX, dst, ELOG, EDEN, 1);
    edge_scatter<<<blocks_for((long long)N_EDGES * (HID / 4), BS), BS, 0, stream>>>(
        F2, ELOG, EDEN, src, dst, O2, 1);
    add_bias_inplace<<<blocks_for((long long)N_NODES * HID, BS), BS, 0, stream>>>(
        O2, b2, (long long)N_NODES * HID, HID);

    // =========== Graph mean pooling ===========
    fill_f32<<<blocks_for(N_GRAPHS * HID, BS), BS, 0, stream>>>(HG, 0.0f, N_GRAPHS * HID);
    fill_f32<<<1, N_GRAPHS, 0, stream>>>(CNT, 0.0f, N_GRAPHS);
    graph_accum<<<blocks_for((long long)N_NODES * (HID / 4), BS), BS, 0, stream>>>(O2, gid, HG, CNT);
    graph_div<<<blocks_for(N_GRAPHS * HID, BS), BS, 0, stream>>>(HG, CNT);

    // =========== Node head ===========
    // T = relu(h2 @ nW1 + nb1)  (40000x128 @ 128x128) fused epilogue
    {
        dim3 grid(N_NODES / 16, 1);
        wmma_gemm_bpk<<<grid, 128, 0, stream>>>(O2, PB3, T, N_NODES, HID, HID, nb1, 1);
    }
    node_head<<<blocks_for(N_NODES, BS), BS, 0, stream>>>(T, nW2, nb2, out);

    // =========== Graph head ===========
    graph_mlp<<<N_GRAPHS, HID, 0, stream>>>(HG, gW1, gb1, gW2, gb2, out + (size_t)N_NODES * 2);
}
